// ResidualAttentionBlock_59871844107180
// MI455X (gfx1250) — matrix operand order repaired and checked by reading, not compiled, not run on hardware
//
#include <hip/hip_runtime.h>
#include <cstdint>

typedef __attribute__((ext_vector_type(16))) _Float16 v16h;
typedef __attribute__((ext_vector_type(8)))  float    v8f;

union HF16 { v16h v; uint4 q[2]; };

__device__ __forceinline__ v8f wmma16(const HF16& a, const HF16& b, v8f c) {
  return __builtin_amdgcn_wmma_f32_16x16x32_f16(false, a.v, false, b.v, (short)0, c, false, false);
}

__device__ __forceinline__ float block_sum(float v, float* sb) {
  int lane = threadIdx.x & 31, w = threadIdx.x >> 5;
#pragma unroll
  for (int m = 16; m; m >>= 1) v += __shfl_xor(v, m, 32);
  if (lane == 0) sb[w] = v;
  __syncthreads();
  int nw = (blockDim.x + 31) >> 5;
  float t = ((int)threadIdx.x < nw) ? sb[threadIdx.x] : 0.f;
  if (w == 0) {
#pragma unroll
    for (int m = 16; m; m >>= 1) t += __shfl_xor(t, m, 32);
    if (lane == 0) sb[0] = t;
  }
  __syncthreads();
  float r = sb[0];
  __syncthreads();
  return r;
}

// ---------------- Generic WMMA GEMM: C[m,n] = act(post*(sum_k A[m,k]*W[n,k] + bias[n])) ----------
// A: f16 row-major (lda), W: f32 row-major (ldw, "w.T" convention), tiles 64x64, K%32==0.
__global__ __launch_bounds__(128) void gemm_f16w(
    const _Float16* __restrict__ A, int lda,
    const float* __restrict__ W, int ldw,
    const float* __restrict__ bias,
    float* __restrict__ Cf, _Float16* __restrict__ Ch, int ldc,
    int K, float post, int act, int accum)
{
  __shared__ _Float16 As[64 * 32];
  __shared__ _Float16 Bs[64 * 32];
  const int tid = threadIdx.x;
  const int lane = tid & 31, wave = tid >> 5;
  const int m0 = blockIdx.x * 64, n0 = blockIdx.y * 64;

  v8f acc[4] = {};
  const int arow = wave * 16 + (lane & 15);
  const int kb = (lane >> 4) * 16;
  const int kbB = (lane >> 4) * 16;

  for (int k0 = 0; k0 < K; k0 += 32) {
    {
      int row = tid >> 1, seg = tid & 1;
      const _Float16* src = A + (size_t)(m0 + row) * lda + k0 + seg * 16;
      uint4* dst = (uint4*)&As[row * 32 + seg * 16];
      dst[0] = ((const uint4*)src)[0];
      dst[1] = ((const uint4*)src)[1];
    }
    {
      int nr = tid >> 1, seg = tid & 1;
      const float* src = W + (size_t)(n0 + nr) * ldw + k0 + seg * 16;
      _Float16* dst = &Bs[nr * 32 + seg * 16];
#pragma unroll
      for (int j = 0; j < 16; j += 4) {
        float4 f = ((const float4*)src)[j >> 2];
        dst[j + 0] = (_Float16)f.x; dst[j + 1] = (_Float16)f.y;
        dst[j + 2] = (_Float16)f.z; dst[j + 3] = (_Float16)f.w;
      }
    }
    __syncthreads();
    HF16 af;
    af.q[0] = *(const uint4*)&As[arow * 32 + kb];
    af.q[1] = *(const uint4*)&As[arow * 32 + kb + 8];
#pragma unroll
    for (int nt = 0; nt < 4; ++nt) {
      HF16 bf;
      const _Float16* bp = &Bs[(nt * 16 + (lane & 15)) * 32 + kbB];
      bf.q[0] = ((const uint4*)bp)[0];
      bf.q[1] = ((const uint4*)bp)[1];
      acc[nt] = wmma16(af, bf, acc[nt]);
    }
    __syncthreads();
  }

  const int nc = lane & 15;
  const int mhi = (lane >> 4) * 8;
#pragma unroll
  for (int nt = 0; nt < 4; ++nt) {
    int col = n0 + nt * 16 + nc;
    float bv = bias ? bias[col] : 0.f;
#pragma unroll
    for (int e = 0; e < 8; ++e) {
      size_t row = (size_t)(m0 + wave * 16 + mhi + e);
      float v = (acc[nt][e] + bv) * post;
      if (act == 1) v = 0.5f * v * (1.f + erff(v * 0.70710678118654752f));       // exact GELU
      else if (act == 2) v = v / (1.f + __expf(-1.702f * v));                   // QuickGELU
      size_t idx = row * (size_t)ldc + col;
      if (accum) v += Cf[idx];
      if (Cf) Cf[idx] = v;
      if (Ch) Ch[idx] = (_Float16)v;
    }
  }
}

// ---------------- LayerNorm: fp32 stats, f16 output ----------------
__global__ __launch_bounds__(256) void ln_kernel(const float* __restrict__ X,
    const float* __restrict__ g, const float* __restrict__ b,
    _Float16* __restrict__ O)
{
  __shared__ float sb[64];
  size_t row = blockIdx.x;
  const float* x = X + row * 768;
  int c = threadIdx.x;
  float v0 = x[c], v1 = x[c + 256], v2 = x[c + 512];
  float mu = block_sum(v0 + v1 + v2, sb) * (1.f / 768.f);
  float d0 = v0 - mu, d1 = v1 - mu, d2 = v2 - mu;
  float var = block_sum(d0 * d0 + d1 * d1 + d2 * d2, sb) * (1.f / 768.f);
  float rstd = rsqrtf(var + 1e-5f);
  O[row * 768 + c]       = (_Float16)(d0 * rstd * g[c] + b[c]);
  O[row * 768 + c + 256] = (_Float16)(d1 * rstd * g[c + 256] + b[c + 256]);
  O[row * 768 + c + 512] = (_Float16)(d2 * rstd * g[c + 512] + b[c + 512]);
}

// ---------------- Fused per-(n,h,qtile) attention, 1 wave ----------------
// Q prescaled by 1/8. KV: k at col 0, v at col 768 (row stride 1536).
__global__ __launch_bounds__(32) void attn_kernel(
    const _Float16* __restrict__ Q, const _Float16* __restrict__ KV,
    _Float16* __restrict__ OH)
{
  __shared__ _Float16 Qt[16 * 64];
  __shared__ _Float16 VtT[64 * 224];
  __shared__ float Sc[16 * 224];
  __shared__ _Float16 Pt[16 * 224];
  __shared__ float invs[16];

  int bi = blockIdx.x;
  int qt = bi % 13;
  int h = (bi / 13) % 12;
  int n = bi / (13 * 12);
  int lane = threadIdx.x;

  { // Q tile 16x64
    int row = lane >> 1, seg = lane & 1;
    int qg = qt * 16 + row; if (qg > 196) qg = 196;
    const uint4* src = (const uint4*)(Q + ((size_t)qg * 64 + n) * 768 + h * 64 + seg * 32);
    uint4* dst = (uint4*)&Qt[row * 64 + seg * 32];
    dst[0] = src[0]; dst[1] = src[1]; dst[2] = src[2]; dst[3] = src[3];
  }
  for (int r = lane; r < 224; r += 32) { // V transposed [hd][tok]
    int tok = r > 196 ? 196 : r;
    const _Float16* vsrc = KV + ((size_t)tok * 64 + n) * 1536 + 768 + h * 64;
    union { uint4 q[8]; _Float16 hh[64]; } vv;
#pragma unroll
    for (int i = 0; i < 8; ++i) vv.q[i] = ((const uint4*)vsrc)[i];
#pragma unroll
    for (int d = 0; d < 64; ++d) VtT[d * 224 + r] = vv.hh[d];
  }
  __syncthreads();

  const int arow = lane & 15;
  const int kb = (lane >> 4) * 16;
  const int kbB = (lane >> 4) * 16;
  for (int kt = 0; kt < 14; ++kt) {
    v8f acc = {};
    int ktok = kt * 16 + (lane & 15);
    int ktokc = ktok > 196 ? 196 : ktok;
    const _Float16* kbase = KV + ((size_t)ktokc * 64 + n) * 1536 + h * 64;
#pragma unroll
    for (int s = 0; s < 2; ++s) {
      HF16 af, bf;
      af.q[0] = *(const uint4*)&Qt[arow * 64 + s * 32 + kb];
      af.q[1] = *(const uint4*)&Qt[arow * 64 + s * 32 + kb + 8];
      bf.q[0] = ((const uint4*)(kbase + s * 32 + kbB))[0];
      bf.q[1] = ((const uint4*)(kbase + s * 32 + kbB))[1];
      acc = wmma16(af, bf, acc);
    }
    int nc = lane & 15, mhi = (lane >> 4) * 8;
    bool kvalid = (kt * 16 + nc) < 197;
#pragma unroll
    for (int e = 0; e < 8; ++e)
      Sc[(mhi + e) * 224 + kt * 16 + nc] = kvalid ? acc[e] : -1e30f;
  }
  __syncthreads();
  if (lane < 16) { // row softmax (unnormalized; 1/sum folded into output)
    float mx = -1e30f;
    for (int c = 0; c < 224; ++c) { float v = Sc[lane * 224 + c]; mx = v > mx ? v : mx; }
    float sum = 0.f;
    for (int c = 0; c < 224; ++c) {
      float e = __expf(Sc[lane * 224 + c] - mx);
      Pt[lane * 224 + c] = (_Float16)e;
      sum += e;
    }
    invs[lane] = 1.f / sum;
  }
  __syncthreads();
#pragma unroll
  for (int nt = 0; nt < 4; ++nt) {
    v8f acc = {};
    int nd = nt * 16 + (lane & 15);
    for (int ks = 0; ks < 7; ++ks) {
      HF16 af, bf;
      af.q[0] = *(const uint4*)&Pt[arow * 224 + ks * 32 + kb];
      af.q[1] = *(const uint4*)&Pt[arow * 224 + ks * 32 + kb + 8];
      bf.q[0] = *(const uint4*)&VtT[nd * 224 + ks * 32 + kbB];
      bf.q[1] = *(const uint4*)&VtT[nd * 224 + ks * 32 + kbB + 8];
      acc = wmma16(af, bf, acc);
    }
    int nc = lane & 15, mhi = (lane >> 4) * 8;
#pragma unroll
    for (int e = 0; e < 8; ++e) {
      int m = mhi + e, qg = qt * 16 + m;
      if (qg < 197)
        OH[((size_t)qg * 64 + n) * 768 + h * 64 + nt * 16 + nc] = (_Float16)(acc[e] * invs[m]);
    }
  }
}

// ---------------- ori_w[n] = sum_{q,k} exp(Qfull.Kfull) (Q prescaled 1/8), WMMA over K=768 ------
__global__ __launch_bounds__(32) void oriw_kernel(const _Float16* __restrict__ Q,
    const _Float16* __restrict__ KV, float* __restrict__ OW)
{
  __shared__ _Float16 Qt[16 * 768];
  int qt = blockIdx.x % 13;
  int n = blockIdx.x / 13;
  int lane = threadIdx.x;
  for (int i = lane; i < 16 * 24; i += 32) {
    int row = i / 24, seg = i % 24;
    int qg = qt * 16 + row; if (qg > 196) qg = 196;
    const uint4* src = (const uint4*)(Q + ((size_t)qg * 64 + n) * 768 + seg * 32);
    uint4* dst = (uint4*)&Qt[row * 768 + seg * 32];
    dst[0] = src[0]; dst[1] = src[1]; dst[2] = src[2]; dst[3] = src[3];
  }
  __syncthreads();
  const int arow = lane & 15, kb = (lane >> 4) * 16, kbB = (lane >> 4) * 16;
  float sum = 0.f;
  for (int kt = 0; kt < 13; ++kt) {
    v8f acc = {};
    int ktok = kt * 16 + (lane & 15);
    int ktokc = ktok > 196 ? 196 : ktok;
    const _Float16* kbase = KV + ((size_t)ktokc * 64 + n) * 1536;
    for (int s = 0; s < 24; ++s) {
      HF16 af, bf;
      af.q[0] = *(const uint4*)&Qt[arow * 768 + s * 32 + kb];
      af.q[1] = *(const uint4*)&Qt[arow * 768 + s * 32 + kb + 8];
      bf.q[0] = ((const uint4*)(kbase + s * 32 + kbB))[0];
      bf.q[1] = ((const uint4*)(kbase + s * 32 + kbB))[1];
      acc = wmma16(af, bf, acc);
    }
    int nc = lane & 15, mhi = (lane >> 4) * 8;
    bool kvalid = (kt * 16 + nc) < 197;
#pragma unroll
    for (int e = 0; e < 8; ++e) {
      int qg = qt * 16 + mhi + e;
      if (kvalid && qg < 197) sum += __expf(acc[e]);
    }
  }
#pragma unroll
  for (int m = 16; m; m >>= 1) sum += __shfl_xor(sum, m, 32);
  if (lane == 0) atomicAdd(OW + n, sum);
}

// ---------------- crs_w[n] = sum_q exp(Q[q,n] . kt[n]) ----------------
__global__ __launch_bounds__(256) void crsw_kernel(const _Float16* __restrict__ Q,
    const _Float16* __restrict__ KVT, float* __restrict__ CW)
{
  __shared__ float ktr[768];
  __shared__ float sb[64];
  int n = blockIdx.x;
  for (int c = threadIdx.x; c < 768; c += 256) ktr[c] = (float)KVT[(size_t)n * 1536 + c];
  __syncthreads();
  float part = 0.f;
  for (int q = threadIdx.x; q < 197; q += 256) {
    const _Float16* qr = Q + ((size_t)q * 64 + n) * 768;
    float s = 0.f;
    for (int d = 0; d < 768; ++d) s += (float)qr[d] * ktr[d];
    part += __expf(s);
  }
  float tot = block_sum(part, sb);
  if (threadIdx.x == 0) CW[n] = tot;
}

// ---------------- temporal attention (T=8 within each of 8 clips, 12 heads) ----------------
__global__ __launch_bounds__(64) void temporal_attn(const _Float16* __restrict__ QT,
    const _Float16* __restrict__ KVT, _Float16* __restrict__ O)
{
  __shared__ float qs[8][64], ks[8][64], vs[8][64], sc[8][8], pr[8][8];
  int bb = blockIdx.x / 12, h = blockIdx.x % 12;
  int d = threadIdx.x;
  for (int t = 0; t < 8; ++t) {
    int j = t * 8 + bb;
    qs[t][d] = (float)QT[(size_t)j * 768 + h * 64 + d];
    ks[t][d] = (float)KVT[(size_t)j * 1536 + h * 64 + d];
    vs[t][d] = (float)KVT[(size_t)j * 1536 + 768 + h * 64 + d];
  }
  __syncthreads();
  {
    int t = d >> 3, u = d & 7;
    float s = 0.f;
    for (int e = 0; e < 64; ++e) s += qs[t][e] * ks[u][e];
    sc[t][u] = s;
  }
  __syncthreads();
  if (d < 8) {
    float mx = -1e30f;
    for (int u = 0; u < 8; ++u) mx = sc[d][u] > mx ? sc[d][u] : mx;
    float sum = 0.f;
    for (int u = 0; u < 8; ++u) { float e = __expf(sc[d][u] - mx); pr[d][u] = e; sum += e; }
    float inv = 1.f / sum;
    for (int u = 0; u < 8; ++u) pr[d][u] *= inv;
  }
  __syncthreads();
  for (int t = 0; t < 8; ++t) {
    float o = 0.f;
    for (int u = 0; u < 8; ++u) o += pr[t][u] * vs[u][d];
    O[(size_t)(t * 8 + bb) * 768 + h * 64 + d] = (_Float16)o;
  }
}

// ---------------- small helpers ----------------
__global__ void zero_kernel(float* p, int n) {
  int i = blockIdx.x * blockDim.x + threadIdx.x;
  if (i < n) p[i] = 0.f;
}
__global__ void gather_ct(const float* __restrict__ x, float* __restrict__ xtr) {
  int j = blockIdx.x; int t = j >> 3, bb = j & 7;
  int src = bb * 8 + t;
  for (int c = threadIdx.x; c < 768; c += blockDim.x)
    xtr[(size_t)j * 768 + c] = x[(size_t)src * 768 + c];
}
__global__ void reorder64(const _Float16* __restrict__ in, _Float16* __restrict__ out) {
  int r = blockIdx.x; int bb = r >> 3, t = r & 7;
  int j = t * 8 + bb;
  for (int c = threadIdx.x; c < 768; c += blockDim.x)
    out[(size_t)r * 768 + c] = in[(size_t)j * 768 + c];
}
__global__ void sain_kernel(const float* __restrict__ crs, const float* __restrict__ ow,
    const float* __restrict__ cw, _Float16* __restrict__ sain) {
  int n = blockIdx.x;
  float lam = cw[n] / (cw[n] + ow[n]);
  for (int c = threadIdx.x; c < 768; c += blockDim.x)
    sain[(size_t)n * 768 + c] = (_Float16)(lam * crs[(size_t)n * 768 + c]);
}
__global__ void combine_kernel(const float* __restrict__ x, const float* __restrict__ oattn,
    const float* __restrict__ sadapt, const float* __restrict__ ow, const float* __restrict__ cw,
    float* __restrict__ x1, size_t total) {
  size_t i = (size_t)blockIdx.x * blockDim.x + threadIdx.x;
  if (i >= total) return;
  int col = (int)(i % 768);
  int n = (int)((i / 768) & 63);
  float lam = cw[n] / (cw[n] + ow[n]);
  x1[i] = x[i] + (1.f - lam) * oattn[i] + sadapt[(size_t)n * 768 + col];
}
__global__ void final_kernel(const float* __restrict__ x1, const float* __restrict__ mlp,
    float* __restrict__ out, size_t total) {
  size_t i = (size_t)blockIdx.x * blockDim.x + threadIdx.x;
  if (i < total) out[i] = x1[i] + mlp[i] + out[i];
}

extern "C" void kernel_launch(void* const* d_in, const int* in_sizes, int n_in,
                              void* d_out, int out_size, void* d_ws, size_t ws_size,
                              hipStream_t stream)
{
  (void)in_sizes; (void)n_in; (void)out_size; (void)ws_size;
  const float* x     = (const float*)d_in[0];
  const float* in_w  = (const float*)d_in[1];
  const float* in_b  = (const float*)d_in[2];
  const float* out_w = (const float*)d_in[3];
  const float* out_b = (const float*)d_in[4];
  const float* ln1_g = (const float*)d_in[5];
  const float* ln1_b = (const float*)d_in[6];
  const float* ln2_g = (const float*)d_in[7];
  const float* ln2_b = (const float*)d_in[8];
  const float* fc_w  = (const float*)d_in[9];
  const float* fc_b  = (const float*)d_in[10];
  const float* proj_w = (const float*)d_in[11];
  const float* proj_b = (const float*)d_in[12];
  const float* sa1_w = (const float*)d_in[13];
  const float* sa1_b = (const float*)d_in[14];
  const float* sa2_w = (const float*)d_in[15];
  const float* sa2_b = (const float*)d_in[16];
  const float* ta1_w = (const float*)d_in[17];
  const float* ta1_b = (const float*)d_in[18];
  const float* ta2_w = (const float*)d_in[19];
  const float* ta2_b = (const float*)d_in[20];
  const float* ma1_w = (const float*)d_in[21];
  const float* ma1_b = (const float*)d_in[22];
  const float* ma2_w = (const float*)d_in[23];
  const float* ma2_b = (const float*)d_in[24];
  float* outp = (float*)d_out;

  const int M = 197 * 64; // 12608
  char* base = (char*)d_ws;
  size_t off = 0;
  auto alloc = [&](size_t bytes) -> void* {
    void* p = base + off;
    off = (off + bytes + 255) & ~(size_t)255;
    return p;
  };

  _Float16* xln16 = (_Float16*)alloc((size_t)M * 768 * 2);
  _Float16* q16   = (_Float16*)alloc((size_t)M * 768 * 2);
  void* regC = alloc((size_t)M * 1536 * 2);           // kv16 (f16) then mlp (f32)
  _Float16* kv16 = (_Float16*)regC; float* mlp = (float*)regC;
  void* regD = alloc((size_t)M * 768 * 2);            // out_heads16 then xn16
  _Float16* outh16 = (_Float16*)regD; _Float16* xn16 = (_Float16*)regD;
  void* regE = alloc((size_t)M * 768 * 4);            // oattn (f32) then h16 (f16, 1536 wide)
  float* oattn = (float*)regE; _Float16* h16 = (_Float16*)regE;
  float* x1 = (float*)alloc((size_t)M * 768 * 4);
  _Float16* t1_16 = (_Float16*)alloc((size_t)M * 192 * 2);
  float* xtraw      = (float*)alloc(64 * 768 * 4);
  _Float16* xtln16  = (_Float16*)alloc(64 * 768 * 2);
  _Float16* qt16    = (_Float16*)alloc(64 * 768 * 2);
  _Float16* kvln16  = (_Float16*)alloc(64 * 1536 * 2);
  _Float16* att_t16 = (_Float16*)alloc(64 * 768 * 2);
  _Float16* xt1_16  = (_Float16*)alloc(64 * 768 * 2);
  _Float16* tta1_16 = (_Float16*)alloc(64 * 192 * 2);
  _Float16* taout16 = (_Float16*)alloc(64 * 768 * 2);
  _Float16* xtok16  = (_Float16*)alloc(64 * 768 * 2);
  _Float16* kvt16   = (_Float16*)alloc(64 * 1536 * 2);
  float* crs_out = (float*)alloc(64 * 768 * 4);
  _Float16* sain16 = (_Float16*)alloc(64 * 768 * 2);
  _Float16* s1_16  = (_Float16*)alloc(64 * 192 * 2);
  float* sadapt = (float*)alloc(64 * 768 * 4);
  float* ow = (float*)alloc(64 * 4);
  float* cw = (float*)alloc(64 * 4);   // contiguous after ow

  const dim3 B128(128);
  const size_t total = (size_t)M * 768;
  const int eb = (int)((total + 255) / 256);

  zero_kernel<<<1, 128, 0, stream>>>(ow, 128); // zero ow + cw

  // main path: LN1 -> Q (prescaled 1/8) and KV
  ln_kernel<<<M, 256, 0, stream>>>(x, ln1_g, ln1_b, xln16);
  gemm_f16w<<<dim3(197, 12), B128, 0, stream>>>(xln16, 768, in_w, 768, in_b,
      nullptr, q16, 768, 768, 0.125f, 0, 0);
  gemm_f16w<<<dim3(197, 24), B128, 0, stream>>>(xln16, 768, in_w + 768 * 768, 768, in_b + 768,
      nullptr, kv16, 1536, 768, 1.f, 0, 0);

  // temporal path on class tokens
  gather_ct<<<64, 256, 0, stream>>>(x, xtraw);
  ln_kernel<<<64, 256, 0, stream>>>(xtraw, ln1_g, ln1_b, xtln16);
  gemm_f16w<<<dim3(1, 12), B128, 0, stream>>>(xtln16, 768, in_w, 768, in_b,
      nullptr, qt16, 768, 768, 0.125f, 0, 0);
  gemm_f16w<<<dim3(1, 24), B128, 0, stream>>>(xtln16, 768, in_w + 768 * 768, 768, in_b + 768,
      nullptr, kvln16, 1536, 768, 1.f, 0, 0);
  temporal_attn<<<96, 64, 0, stream>>>(qt16, kvln16, att_t16);
  gemm_f16w<<<dim3(1, 12), B128, 0, stream>>>(att_t16, 768, out_w, 768, out_b,
      nullptr, xt1_16, 768, 768, 1.f, 0, 0);
  gemm_f16w<<<dim3(1, 3), B128, 0, stream>>>(xt1_16, 768, ta1_w, 768, ta1_b,
      nullptr, tta1_16, 192, 768, 1.f, 1, 0);
  gemm_f16w<<<dim3(1, 12), B128, 0, stream>>>(tta1_16, 192, ta2_w, 192, ta2_b,
      nullptr, taout16, 768, 192, 1.f, 0, 0);
  reorder64<<<64, 256, 0, stream>>>(taout16, xtok16);
  gemm_f16w<<<dim3(1, 24), B128, 0, stream>>>(xtok16, 768, in_w + 768 * 768, 768, in_b + 768,
      nullptr, kvt16, 1536, 768, 1.f, 0, 0);
  gemm_f16w<<<dim3(1, 12), B128, 0, stream>>>(kvt16 + 768, 1536, out_w, 768, out_b,
      crs_out, nullptr, 768, 768, 1.f, 0, 0);

  // attention-weight scalars + per-head attention + out-proj
  oriw_kernel<<<64 * 13, 32, 0, stream>>>(q16, kv16, ow);
  crsw_kernel<<<64, 256, 0, stream>>>(q16, kvt16, cw);
  attn_kernel<<<64 * 12 * 13, 32, 0, stream>>>(q16, kv16, outh16);
  gemm_f16w<<<dim3(197, 12), B128, 0, stream>>>(outh16, 768, out_w, 768, out_b,
      oattn, nullptr, 768, 768, 1.f, 0, 0);

  // sa adapter on 64 distinct rows, then combine
  sain_kernel<<<64, 256, 0, stream>>>(crs_out, ow, cw, sain16);
  gemm_f16w<<<dim3(1, 3), B128, 0, stream>>>(sain16, 768, sa1_w, 768, sa1_b,
      nullptr, s1_16, 192, 768, 1.f, 1, 0);
  gemm_f16w<<<dim3(1, 12), B128, 0, stream>>>(s1_16, 192, sa2_w, 192, sa2_b,
      sadapt, nullptr, 768, 192, 1.f, 0, 0);
  combine_kernel<<<eb, 256, 0, stream>>>(x, oattn, sadapt, ow, cw, x1, total);

  // MLP (QuickGELU) in two 1536-col chunks + ma adapter
  ln_kernel<<<M, 256, 0, stream>>>(x1, ln2_g, ln2_b, xn16);
  gemm_f16w<<<dim3(197, 24), B128, 0, stream>>>(xn16, 768, fc_w, 768, fc_b,
      nullptr, h16, 1536, 768, 1.f, 2, 0);
  gemm_f16w<<<dim3(197, 12), B128, 0, stream>>>(h16, 1536, proj_w, 3072, proj_b,
      mlp, nullptr, 768, 1536, 1.f, 0, 0);
  gemm_f16w<<<dim3(197, 24), B128, 0, stream>>>(xn16, 768, fc_w + 1536 * 768, 768, fc_b + 1536,
      nullptr, h16, 1536, 768, 1.f, 2, 0);
  gemm_f16w<<<dim3(197, 12), B128, 0, stream>>>(h16, 1536, proj_w + 1536, 3072, nullptr,
      mlp, nullptr, 768, 1536, 1.f, 0, 1);
  gemm_f16w<<<dim3(197, 3), B128, 0, stream>>>(xn16, 768, ma1_w, 768, ma1_b,
      nullptr, t1_16, 192, 768, 1.f, 1, 0);
  gemm_f16w<<<dim3(197, 12), B128, 0, stream>>>(t1_16, 192, ma2_w, 192, ma2_b,
      outp, nullptr, 768, 192, 1.f, 0, 0);

  final_kernel<<<eb, 256, 0, stream>>>(x1, mlp, outp, total);
}
